// FourierBasis_12893491823264
// MI455X (gfx1250) — compile-verified, matrix operand order checked by reading, not run on hardware
//
#include <hip/hip_runtime.h>
#include <hip/hip_bf16.h>

// ---------------------------------------------------------------------------
// FourierBasis neural-BRDF kernel for MI455X (gfx1250, wave32, WMMA).
//
// Compute-bound MLP (63->128, 5x (191->128 skip), 128->48) mapped to
// v_wmma_f32_16x16x32_f16.  Weights pre-swizzled to f16 B-fragment layout in
// d_ws; activations live in LDS as row-major f16 with skip concat [h | enc].
// Rolled K-loops bound VGPR pressure (<256).  Inside each K-step, a
// sched_barrier separates the 16-load B clause from the 8 WMMAs so all
// fragments stay in flight and WMMAs drain behind staggered loadcnt waits.
// ---------------------------------------------------------------------------

typedef __attribute__((ext_vector_type(16))) _Float16 v16h;
typedef __attribute__((ext_vector_type(8)))  float    v8f;

#define NPTS    262144
#define NFREQ   10
#define ORDER   16
#define HIDDEN  128
#define NHL     5
#define ENC     63      // actual encoding dim
#define ACTC    192     // activation cols: 128 h + 64 padded enc
#define OUTF    48      // 3 * ORDER
#define ROWS    256     // points per workgroup
#define NEG_SLOPE 0.01f

// Fragment layout in the pre-swizzled f16 weight buffer (frag = 512 halves):
//   W0  : 2 ksteps * 8 ntiles            = 16 frags   (offset 0)
//   Wh  : 5 layers * 6 ksteps * 8 ntiles = 240 frags  (offset 16)
//   Wout: 4 ksteps * 3 ntiles            = 12 frags   (offset 256)
#define FRAG_W0_BASE   0
#define FRAG_WH_BASE   16
#define FRAG_WOUT_BASE 256
#define FRAG_TOTAL     268
#define WF16_ELEMS     (FRAG_TOTAL * 512)

// ---------------------------------------------------------------------------
// Prep: fp32 weights -> f16 swizzled into the 16x16x32 f16 WMMA B-matrix
// per-lane layout (mirror of the documented 16-bit A layout, ISA 7.12.2):
//   lane: N=lane&15, K-half = lane>=16; element e (v=e/2, h=e&1):
//   k = ((v&3)*2 + h) + (lane>=16 ? 8 : 0) + (v>=4 ? 16 : 0)
// Main kernel then fetches a whole B fragment as one 32-byte load per lane.
// ---------------------------------------------------------------------------
__global__ void prep_weights_kernel(const float* __restrict__ W0,
                                    const float* __restrict__ Wh,
                                    const float* __restrict__ Wout,
                                    _Float16* __restrict__ wf) {
    int gid = blockIdx.x * 256 + threadIdx.x;
    if (gid >= WF16_ELEMS) return;
    int frag = gid >> 9;
    int slot = gid & 511;
    int lane = slot >> 4;
    int e    = slot & 15;
    int v = e >> 1, h = e & 1;
    int kidx = ((v & 3) * 2 + h) + ((lane & 16) ? 8 : 0) + ((v & 4) ? 16 : 0);
    int ncol = lane & 15;

    float val = 0.0f;
    if (frag < FRAG_WH_BASE) {                       // W0: (63,128), K padded to 64
        int ks = frag >> 3, nt = frag & 7;
        int k = ks * 32 + kidx, n = nt * 16 + ncol;
        if (k < ENC) val = W0[k * HIDDEN + n];
    } else if (frag < FRAG_WOUT_BASE) {              // Wh: (5,191,128), K padded to 192
        int f = frag - FRAG_WH_BASE;
        int layer = f / 48, rem = f % 48;
        int ks = rem >> 3, nt = rem & 7;
        int k = ks * 32 + kidx, n = nt * 16 + ncol;
        if (k < HIDDEN + ENC) val = Wh[(layer * (HIDDEN + ENC) + k) * HIDDEN + n];
    } else {                                         // Wout: (128,48)
        int f = frag - FRAG_WOUT_BASE;
        int ks = f / 3, nt = f % 3;
        int k = ks * 32 + kidx, n = nt * 16 + ncol;
        val = Wout[k * OUTF + n];
    }
    wf[gid] = (_Float16)val;
}

// ---------------------------------------------------------------------------
// Scalar geometry helpers (fp32, matches reference semantics).
// ---------------------------------------------------------------------------
__device__ __forceinline__ void norm3(float& x, float& y, float& z, float eps) {
    float inv = 1.0f / fmaxf(sqrtf(x * x + y * y + z * z), eps);
    x *= inv; y *= inv; z *= inv;
}
__device__ __forceinline__ void cross3(float ax, float ay, float az,
                                       float bx, float by, float bz,
                                       float& cx, float& cy, float& cz) {
    cx = ay * bz - az * by;
    cy = az * bx - ax * bz;
    cz = ax * by - ay * bx;
}

// ---------------------------------------------------------------------------
// Main fused kernel: 256 threads = 8 waves; each wave owns 32 rows (2 M-tiles).
// LDS: act [ROWS][192] f16 | cheb [ROWS][16] f32 | prod [ROWS][48] f32 = 160 KB.
// ---------------------------------------------------------------------------
__global__ __launch_bounds__(256)
void fourier_mlp_kernel(const float* __restrict__ xin,
                        const float* __restrict__ view,
                        const float* __restrict__ normal,
                        const float* __restrict__ light,
                        const float* __restrict__ b0,
                        const float* __restrict__ bh,
                        const float* __restrict__ bout,
                        const _Float16* __restrict__ wf16,
                        float* __restrict__ out) {
    extern __shared__ char smem[];
    _Float16* act  = (_Float16*)smem;                          // ROWS*ACTC f16
    float*    chb  = (float*)(smem + ROWS * ACTC * 2);         // ROWS*16 f32
    float*    prod = chb + ROWS * 16;                          // ROWS*48 f32

    const int t = threadIdx.x;
    const int p = blockIdx.x * ROWS + t;

    // ---------------- per-point encoding + geometry (fp32 scalar) -----------
    {
        float x0 = xin[p * 3 + 0], x1 = xin[p * 3 + 1], x2 = xin[p * 3 + 2];
        _Float16* erow = act + t * ACTC + HIDDEN;   // enc lives at cols 128..191
        erow[0] = (_Float16)x0; erow[1] = (_Float16)x1; erow[2] = (_Float16)x2;
        float xc[3] = {x0, x1, x2};
        float freq = 1.0f;
        #pragma unroll
        for (int f = 0; f < NFREQ; ++f) {
            #pragma unroll
            for (int c = 0; c < 3; ++c) {
                float ang = xc[c] * freq;
                // hw v_sin/v_cos: error ~1e-5, below the f16 quantization of enc
                erow[3 + f * 3 + c]              = (_Float16)__sinf(ang);
                erow[3 + 3 * NFREQ + f * 3 + c]  = (_Float16)__cosf(ang);
            }
            freq *= 2.0f;
        }
        erow[ENC] = (_Float16)0.0f;                 // pad col 191

        // coordinate_system(normal)
        float nx = normal[p * 3 + 0], ny = normal[p * 3 + 1], nz = normal[p * 3 + 2];
        norm3(nx, ny, nz, 1e-6f);
        float sign = (nz >= 0.0f) ? 1.0f : -1.0f;
        float s_z  = sign + nz;
        float safe = (fabsf(s_z) < 1e-6f) ? copysignf(1e-6f, s_z) : s_z;
        float a = -1.0f / safe;
        float b = nx * ny * a;
        float sx = nx * nx * a * sign + 1.0f, sy = b * sign, sz = -nx * sign;
        norm3(sx, sy, sz, 1e-6f);
        float tx, ty, tz;
        cross3(sx, sy, sz, nx, ny, nz, tx, ty, tz);  norm3(tx, ty, tz, 1e-6f);
        cross3(nx, ny, nz, tx, ty, tz, sx, sy, sz);  norm3(sx, sy, sz, 1e-6f);

        // wo = to_local(frame, normalize(view)); wi = to_local(frame, light)
        float vx = view[p * 3 + 0], vy = view[p * 3 + 1], vz = view[p * 3 + 2];
        norm3(vx, vy, vz, 1e-6f);
        float wox = vx * sx + vy * sy + vz * sz;
        float woy = vx * tx + vy * ty + vz * tz;
        float woz = vx * nx + vy * ny + vz * nz;
        norm3(wox, woy, woz, 1e-7f);
        float lx = light[p * 3 + 0], ly = light[p * 3 + 1], lz = light[p * 3 + 2];
        float wix = lx * sx + ly * sy + lz * sz;
        float wiy = lx * tx + ly * ty + lz * tz;
        float wiz = lx * nx + ly * ny + lz * nz;
        norm3(wix, wiy, wiz, 1e-7f);

        // cos_D_phi(-wi, wo)
        float num = -(wix * wox + wiy * woy);
        float den = sqrtf((wix * wix + wiy * wiy) * (wox * wox + woy * woy));
        float cphi = fminf(fmaxf(num / den, -1.0f), 1.0f);

        // Chebyshev T0..T15
        float* crow = chb + t * 16;
        float tm2 = 1.0f, tm1 = cphi;
        crow[0] = tm2; crow[1] = tm1;
        #pragma unroll
        for (int k = 2; k < ORDER; ++k) {
            float tk = 2.0f * cphi * tm1 - tm2;
            crow[k] = tk;
            tm2 = tm1; tm1 = tk;
        }
    }
    __syncthreads();

    // ---------------- WMMA MLP: each wave -> 2 M-tiles of 16 rows -----------
    const int wave  = t >> 5;
    const int lane  = t & 31;
    const int lhalf = (lane >> 4) & 1;     // 0: lanes 0-15, 1: lanes 16-31
    const int koff  = lhalf * 8;           // K-half select inside A/B fragments
    const int ncol  = lane & 15;           // N column (B/C) / M row (A)
    const v16h* wv  = (const v16h*)wf16;   // 32 v16h per fragment (one per lane)

    #pragma unroll 1
    for (int m2 = 0; m2 < 2; ++m2) {
        const int tb = wave * 32 + m2 * 16;                     // tile row base
        const _Float16* arow = act + (tb + ncol) * ACTC;        // this lane's A row

        // A-fragment: two 16-byte LDS loads (ds_load_b128) per the ISA layout.
        auto loadA = [&](int kbase) -> v16h {
            union { uint4 q[2]; v16h v; } u;
            u.q[0] = *(const uint4*)(arow + kbase + koff);
            u.q[1] = *(const uint4*)(arow + kbase + 16 + koff);
            return u.v;
        };
        // Bias + leaky-ReLU (max(x, 0.01x)) + f16 convert, scatter C tile back.
        auto storeActs = [&](const v8f* acc2, const float* bias) {
            #pragma unroll
            for (int nt = 0; nt < 8; ++nt) {
                float bv = bias[nt * 16 + ncol];
                #pragma unroll
                for (int r = 0; r < 8; ++r) {
                    float vv = acc2[nt][r] + bv;
                    vv = fmaxf(vv, NEG_SLOPE * vv);             // leaky relu, slope<1
                    int row = tb + lhalf * 8 + r;               // C layout (ISA 7.12.2)
                    act[row * ACTC + nt * 16 + ncol] = (_Float16)vv;
                }
            }
        };

        v8f acc[8];
        const v8f vzero = {0.f, 0.f, 0.f, 0.f, 0.f, 0.f, 0.f, 0.f};

        // ---- layer 0: enc(64) -> 128 --------------------------------------
        #pragma unroll
        for (int nt = 0; nt < 8; ++nt) acc[nt] = vzero;
        #pragma unroll 1
        for (int ks = 0; ks < 2; ++ks) {
            v16h a = loadA(HIDDEN + ks * 32);                   // enc cols 128..191
            v16h bm[8];
            #pragma unroll
            for (int nt = 0; nt < 8; ++nt)                      // one load clause
                bm[nt] = wv[(FRAG_W0_BASE + ks * 8 + nt) * 32 + lane];
            __builtin_amdgcn_sched_barrier(0);                  // keep 8 frags live
            #pragma unroll
            for (int nt = 0; nt < 8; ++nt)                      // staggered drain
                acc[nt] = __builtin_amdgcn_wmma_f32_16x16x32_f16(
                    false, a, false, bm[nt], (short)0, acc[nt], false, false);
            __builtin_amdgcn_sched_barrier(0);                  // no cross-iter hoist
        }
        storeActs(acc, b0);
        __builtin_amdgcn_sched_barrier(0);

        // ---- hidden layers: [h|enc](192) -> 128 ---------------------------
        #pragma unroll 1
        for (int layer = 0; layer < NHL; ++layer) {
            #pragma unroll
            for (int nt = 0; nt < 8; ++nt) acc[nt] = vzero;
            #pragma unroll 1
            for (int ks = 0; ks < 6; ++ks) {
                v16h a = loadA(ks * 32);
                v16h bm[8];
                #pragma unroll
                for (int nt = 0; nt < 8; ++nt)
                    bm[nt] = wv[(FRAG_WH_BASE + layer * 48 + ks * 8 + nt) * 32 + lane];
                __builtin_amdgcn_sched_barrier(0);              // keep 8 frags live
                #pragma unroll
                for (int nt = 0; nt < 8; ++nt)
                    acc[nt] = __builtin_amdgcn_wmma_f32_16x16x32_f16(
                        false, a, false, bm[nt], (short)0, acc[nt], false, false);
                __builtin_amdgcn_sched_barrier(0);
            }
            storeActs(acc, bh + layer * HIDDEN);
            __builtin_amdgcn_sched_barrier(0);
        }

        // ---- output layer: 128 -> 48, fold in Chebyshev basis -------------
        v8f oc[3];
        #pragma unroll
        for (int nt = 0; nt < 3; ++nt) oc[nt] = vzero;
        #pragma unroll 1
        for (int ks = 0; ks < 4; ++ks) {
            v16h a = loadA(ks * 32);                            // h cols only
            v16h bm[3];
            #pragma unroll
            for (int nt = 0; nt < 3; ++nt)
                bm[nt] = wv[(FRAG_WOUT_BASE + ks * 3 + nt) * 32 + lane];
            __builtin_amdgcn_sched_barrier(0);                  // keep 3 frags live
            #pragma unroll
            for (int nt = 0; nt < 3; ++nt)
                oc[nt] = __builtin_amdgcn_wmma_f32_16x16x32_f16(
                    false, a, false, bm[nt], (short)0, oc[nt], false, false);
            __builtin_amdgcn_sched_barrier(0);
        }
        #pragma unroll
        for (int nt = 0; nt < 3; ++nt) {
            float bv = bout[nt * 16 + ncol];
            #pragma unroll
            for (int r = 0; r < 8; ++r) {
                int row = tb + lhalf * 8 + r;
                float coeff = oc[nt][r] + bv;
                // col % 16 == ncol -> this lane always multiplies T_{ncol}
                prod[row * OUTF + nt * 16 + ncol] = coeff * chb[row * 16 + ncol];
            }
        }
        __builtin_amdgcn_sched_barrier(0);
    }
    __syncthreads();

    // ---------------- final reduction: rgb[f] = sum_j coeff*T_j -------------
    {
        const float* prow = prod + t * OUTF;
        float r0 = 0.f, r1 = 0.f, r2 = 0.f;
        #pragma unroll
        for (int j = 0; j < 16; ++j) {
            r0 += prow[j];
            r1 += prow[16 + j];
            r2 += prow[32 + j];
        }
        out[p * 3 + 0] = r0;
        out[p * 3 + 1] = r1;
        out[p * 3 + 2] = r2;
    }
}

// ---------------------------------------------------------------------------
extern "C" void kernel_launch(void* const* d_in, const int* in_sizes, int n_in,
                              void* d_out, int out_size, void* d_ws, size_t ws_size,
                              hipStream_t stream) {
    const float* xin    = (const float*)d_in[0];
    const float* view   = (const float*)d_in[1];
    const float* normal = (const float*)d_in[2];
    const float* light  = (const float*)d_in[3];
    const float* W0     = (const float*)d_in[4];
    const float* b0     = (const float*)d_in[5];
    const float* Wh     = (const float*)d_in[6];
    const float* bh     = (const float*)d_in[7];
    const float* Wout   = (const float*)d_in[8];
    const float* bout   = (const float*)d_in[9];
    float* out          = (float*)d_out;
    _Float16* wf16      = (_Float16*)d_ws;      // 274,432 bytes of scratch

    // 1) swizzle weights into WMMA B-fragment layout (f16)
    prep_weights_kernel<<<(WF16_ELEMS + 255) / 256, 256, 0, stream>>>(W0, Wh, Wout, wf16);

    // 2) fused encode + geometry + WMMA MLP + Chebyshev contraction
    const size_t lds = (size_t)ROWS * ACTC * 2 + (size_t)ROWS * 16 * 4 + (size_t)ROWS * OUTF * 4; // 160 KB
    (void)hipFuncSetAttribute((const void*)fourier_mlp_kernel,
                              hipFuncAttributeMaxDynamicSharedMemorySize, (int)lds);
    fourier_mlp_kernel<<<NPTS / ROWS, 256, lds, stream>>>(
        xin, view, normal, light, b0, bh, bout, wf16, out);
}
